// Adapt_gen_85040352460818
// MI455X (gfx1250) — matrix operand order repaired and checked by reading, not compiled, not run on hardware
//
#include <hip/hip_runtime.h>
#include <hip/hip_bf16.h>
#include <math.h>

// ---------------- model constants ----------------
#define BB 8
#define NN 2048
#define CC 3
#define EE 256
#define HH 8
#define NBLK 3
#define KK 32
#define BN (BB*NN)          // 16384 tokens
#define EPSN 1e-5f

typedef __attribute__((ext_vector_type(16))) __bf16 v16bf;
typedef __attribute__((ext_vector_type(8)))  float  v8f;

static __device__ __forceinline__ __bf16 tobf(float x) { return (__bf16)x; }

// pack a float4 into 4 bf16 elements of a fragment
#define CVT4(frag, base, f4) do { \
    (frag)[(base)+0] = tobf((f4).x); (frag)[(base)+1] = tobf((f4).y); \
    (frag)[(base)+2] = tobf((f4).z); (frag)[(base)+3] = tobf((f4).w); } while (0)

// =====================================================================
// KNN: brute force, LDS-tiled candidates, register-resident sorted top-K
// =====================================================================
#define KNN_BLK 128
__global__ void __launch_bounds__(KNN_BLK) k_knn(const float* __restrict__ x,
                                                 int* __restrict__ idx) {
  int q = blockIdx.x * blockDim.x + threadIdx.x;   // 0..B*N-1
  int b = q >> 11;
  int n = q & (NN - 1);
  const float* xb = x + (long)b * NN * 3;
  float qx = xb[n*3+0], qy = xb[n*3+1], qz = xb[n*3+2];
  float dist[KK]; int nbr[KK];
#pragma unroll
  for (int j = 0; j < KK; ++j) { dist[j] = 3.4e38f; nbr[j] = 0; }
  __shared__ float sp[KNN_BLK * 3];
  for (int t0 = 0; t0 < NN; t0 += KNN_BLK) {
    __syncthreads();
    for (int j = threadIdx.x; j < KNN_BLK * 3; j += blockDim.x) sp[j] = xb[t0*3 + j];
    __syncthreads();
    for (int t = 0; t < KNN_BLK; ++t) {
      float dx = qx - sp[t*3+0], dy = qy - sp[t*3+1], dz = qz - sp[t*3+2];
      float d = dx*dx + dy*dy + dz*dz;
      if (d < dist[KK-1]) {
        dist[KK-1] = d; nbr[KK-1] = t0 + t;
#pragma unroll
        for (int j = KK-1; j > 0; --j) {
          if (dist[j] < dist[j-1]) {
            float td = dist[j]; dist[j] = dist[j-1]; dist[j-1] = td;
            int   ti = nbr[j];  nbr[j]  = nbr[j-1];  nbr[j-1]  = ti;
          }
        }
      }
    }
  }
  int base = q * KK;
#pragma unroll
  for (int j = 0; j < KK; ++j) idx[base + j] = nbr[j];
}

// =====================================================================
// ARPE stage 1: feat=[x, x-knn] @ W1 + b1 -> h1 [B,N,K,6]
// =====================================================================
__global__ void __launch_bounds__(256) k_arpe1(const float* __restrict__ x,
                                               const int* __restrict__ idx,
                                               const float* __restrict__ W1,
                                               const float* __restrict__ b1,
                                               float* __restrict__ h1) {
  int t = blockIdx.x * blockDim.x + threadIdx.x;    // B*N*K
  if (t >= BN * KK) return;
  int bn = t >> 5;
  int b = bn >> 11; int n = bn & (NN - 1);
  const float* xb = x + (long)b * NN * 3;
  float f[6];
  f[0] = xb[n*3+0]; f[1] = xb[n*3+1]; f[2] = xb[n*3+2];
  int j = idx[t];
  f[3] = f[0] - xb[j*3+0]; f[4] = f[1] - xb[j*3+1]; f[5] = f[2] - xb[j*3+2];
  float* out = h1 + (long)t * 6;
#pragma unroll
  for (int e = 0; e < 6; ++e) {
    float a = b1[e];
#pragma unroll
    for (int c = 0; c < 6; ++c) a += f[c] * W1[c*6 + e];
    out[e] = a;
  }
}

// per-column sum / sumsq reduction (BatchNorm batch stats), atomics into zeroed buffers
__global__ void __launch_bounds__(64) k_colstats(const float* __restrict__ a,
                                                 int rows, int cols,
                                                 float* __restrict__ sum,
                                                 float* __restrict__ sumsq) {
  int c = blockIdx.y * blockDim.x + threadIdx.x;
  if (c >= cols) return;
  float s = 0.f, q2 = 0.f;
  for (int r = blockIdx.x; r < rows; r += gridDim.x) {
    float v = a[(long)r * cols + c];
    s += v; q2 += v * v;
  }
  atomicAdd(&sum[c], s);
  atomicAdd(&sumsq[c], q2);
}

// =====================================================================
// ARPE stage 2: BN1+ELU, max over K, @W2 + b2 -> h2 [B,N,256]
// =====================================================================
__global__ void __launch_bounds__(256) k_arpe2(const float* __restrict__ h1,
                                               const float* __restrict__ sum1,
                                               const float* __restrict__ sq1,
                                               const float* __restrict__ g1,
                                               const float* __restrict__ be1,
                                               const float* __restrict__ W2,
                                               const float* __restrict__ b2,
                                               float* __restrict__ h2) {
  int bn = blockIdx.x; int tid = threadIdx.x;
  __shared__ float sh[KK][6];
  __shared__ float shp[6];
  if (tid < KK * 6) {
    int k = tid / 6, f = tid % 6;
    float cnt = (float)(BN * KK);
    float mu = sum1[f] / cnt;
    float var = sq1[f] / cnt - mu * mu;
    float v = h1[((long)bn * KK + k) * 6 + f];
    v = g1[f] * (v - mu) * rsqrtf(var + EPSN) + be1[f];
    sh[k][f] = v > 0.f ? v : (expf(v) - 1.f);          // ELU
  }
  __syncthreads();
  if (tid < 6) {
    float m = sh[0][tid];
    for (int k = 1; k < KK; ++k) m = fmaxf(m, sh[k][tid]);
    shp[tid] = m;
  }
  __syncthreads();
  float a = b2[tid];
#pragma unroll
  for (int f = 0; f < 6; ++f) a += shp[f] * W2[f*EE + tid];
  h2[(long)bn * EE + tid] = a;
}

// ARPE stage 3: BN2 + ELU elementwise -> x0
__global__ void __launch_bounds__(256) k_arpe3(const float* __restrict__ h2,
                                               const float* __restrict__ sum2,
                                               const float* __restrict__ sq2,
                                               const float* __restrict__ g2,
                                               const float* __restrict__ be2,
                                               float* __restrict__ xo) {
  int t = blockIdx.x * blockDim.x + threadIdx.x;
  if (t >= BN * EE) return;
  int c = t & (EE - 1);
  float cnt = (float)BN;
  float mu = sum2[c] / cnt;
  float var = sq2[c] / cnt - mu * mu;
  float v = h2[t];
  v = g2[c] * (v - mu) * rsqrtf(var + EPSN) + be2[c];
  xo[t] = v > 0.f ? v : (expf(v) - 1.f);
}

// =====================================================================
// LayerNorm over E=256, one block per token
// =====================================================================
__global__ void __launch_bounds__(256) k_ln(const float* __restrict__ xin,
                                            const float* __restrict__ g,
                                            const float* __restrict__ bb,
                                            float* __restrict__ out) {
  int row = blockIdx.x; int c = threadIdx.x;
  float v = xin[(long)row * EE + c];
  __shared__ float red[256];
  __shared__ float mu_s, rv_s;
  red[c] = v; __syncthreads();
  for (int s = 128; s > 0; s >>= 1) { if (c < s) red[c] += red[c + s]; __syncthreads(); }
  if (c == 0) mu_s = red[0] * (1.f / 256.f);
  __syncthreads();
  float d = v - mu_s;
  red[c] = d * d; __syncthreads();
  for (int s = 128; s > 0; s >>= 1) { if (c < s) red[c] += red[c + s]; __syncthreads(); }
  if (c == 0) rv_s = rsqrtf(red[0] * (1.f / 256.f) + EPSN);
  __syncthreads();
  out[(long)row * EE + c] = g[c] * d * rv_s + bb[c];
}

// =====================================================================
// Generic bf16 WMMA GEMM:  C = act(A[M,Kd] @ W[Kd,Nc] + bias (+ R))
// One 16x64 output tile per wave (4 accumulators) -> one A fragment
// feeds 4 WMMAs. A fragment loads are float4 (global_load_b128).
// A frag: lane=M, elems 0-7 / 8-15 = K k0+half*8..+7 / +16..+23 (ISA layout)
// B frag: lane=N (n0+l), elems = K in the same order as the A frag
// C frag: lane=N, elem r -> M = m0 + half*8 + r
// =====================================================================
template <int ACT>   // 0 = none, 1 = exact GELU
__global__ void __launch_bounds__(256) k_gemm(const float* __restrict__ A,
                                              const float* __restrict__ Wt,
                                              const float* __restrict__ bias,
                                              const float* __restrict__ Rres,
                                              float* __restrict__ Cc,
                                              int M, int Kd, int Nc) {
  int lane = threadIdx.x & 31;
  int wave = threadIdx.x >> 5;
  int tilesN = Nc >> 6;                     // 64-wide output tiles
  int tilesM = (M + 15) >> 4;
  int tile = blockIdx.x * 8 + wave;
  if (tile >= tilesM * tilesN) return;
  int tm = tile / tilesN, tn = tile - tm * tilesN;
  int m0 = tm << 4, n0 = tn << 6;
  int half = lane >> 4, l = lane & 15;

  v8f acc[4] = {{}, {}, {}, {}};
  int arow = m0 + l;
  bool aok = arow < M;
  const float* Abase = A + (long)arow * Kd;

  for (int k0 = 0; k0 < Kd; k0 += 32) {
    // ---- A fragment: two contiguous 8-float runs -> 4x b128 ----
    v16bf af;
    {
      const float4* ap = (const float4*)(Abase + k0 + half * 8);
      float4 a0, a1, a2, a3;
      if (aok) { a0 = ap[0]; a1 = ap[1]; a2 = ap[4]; a3 = ap[5]; }
      else { a0 = a1 = a2 = a3 = make_float4(0.f, 0.f, 0.f, 0.f); }
      CVT4(af, 0, a0); CVT4(af, 4, a1); CVT4(af, 8, a2); CVT4(af, 12, a3);
    }
    const float* brow = Wt + (long)(k0 + half * 8) * Nc + n0 + l;
    if (k0 + 32 < Kd) {                     // prefetch next K-slice
      __builtin_prefetch(Abase + k0 + 32, 0, 1);
      __builtin_prefetch(brow + 32 * Nc, 0, 1);
    }
    // ---- 4 B fragments (lane = N, elems = K strided by Nc) ----
#pragma unroll
    for (int s = 0; s < 4; ++s) {
      const float* bp = brow + s * 16;
      v16bf bfm;
#pragma unroll
      for (int e = 0; e < 8; ++e) { bfm[e] = tobf(bp[(long)e * Nc]); bfm[8 + e] = tobf(bp[(long)(16 + e) * Nc]); }
      acc[s] = __builtin_amdgcn_wmma_f32_16x16x32_bf16(false, af, false, bfm,
                                                       (short)0, acc[s], false, false);
    }
  }

#pragma unroll
  for (int s = 0; s < 4; ++s) {
    int col = n0 + s * 16 + l;
    float bv = bias ? bias[col] : 0.f;
#pragma unroll
    for (int r = 0; r < 8; ++r) {
      int row = m0 + half * 8 + r;
      if (row < M) {
        float v = acc[s][r] + bv;
        if (Rres) v += Rres[(long)row * Nc + col];
        if (ACT == 1) v = 0.5f * v * (1.f + erff(v * 0.70710678118f));
        Cc[(long)row * Nc + col] = v;
      }
    }
  }
}

// =====================================================================
// Flash attention: one wave per (b, h, 16-query tile); dh=32, N=2048
// S = (Q*scale) K^T via WMMA; streaming softmax in C-fragments;
// P transposed C-layout->A-layout via per-wave LDS; O += P V via WMMA
// =====================================================================
__global__ void __launch_bounds__(32) k_attn(const float* __restrict__ qkv,
                                             float* __restrict__ o) {
  int qt = blockIdx.x & 127;             // N/16 = 128 query tiles
  int bh = blockIdx.x >> 7;
  int h = bh & (HH - 1);
  int b = bh >> 3;
  int lane = threadIdx.x;
  int half = lane >> 4, l = lane & 15;
  const float scale = 0.17677669529663687f;   // 1/sqrt(32)

  v16bf aq;
  {
    int row = qt * 16 + l;
    const float4* p = (const float4*)(qkv + ((long)(b * NN + row) * (3 * EE)) + h * 32 + half * 8);
    float4 q0 = p[0], q1 = p[1], q2 = p[4], q3 = p[5];
    q0.x *= scale; q0.y *= scale; q0.z *= scale; q0.w *= scale;
    q1.x *= scale; q1.y *= scale; q1.z *= scale; q1.w *= scale;
    q2.x *= scale; q2.y *= scale; q2.z *= scale; q2.w *= scale;
    q3.x *= scale; q3.y *= scale; q3.z *= scale; q3.w *= scale;
    CVT4(aq, 0, q0); CVT4(aq, 4, q1); CVT4(aq, 8, q2); CVT4(aq, 12, q3);
  }

  v8f o0 = {}, o1 = {};
  float mrow[8], lrow[8];
#pragma unroll
  for (int r = 0; r < 8; ++r) { mrow[r] = -3.4e38f; lrow[r] = 0.f; }

  __shared__ float plds[16][33];   // P tile, padded

  for (int kt = 0; kt < NN; kt += 32) {
    // K fragments: lane = key kt+l / kt+16+l, elems = d in the same order as aq
    v16bf bk0, bk1;
    const float* kb = qkv + ((long)(b * NN + kt + l) * (3 * EE)) + EE + h * 32 + half * 8;
#pragma unroll
    for (int e = 0; e < 8; ++e) { bk0[e] = tobf(kb[e]); bk0[8 + e] = tobf(kb[16 + e]); }
#pragma unroll
    for (int e = 0; e < 8; ++e) { bk1[e] = tobf(kb[(long)16 * (3 * EE) + e]); bk1[8 + e] = tobf(kb[(long)16 * (3 * EE) + 16 + e]); }
    v8f z = {};
    v8f s0 = __builtin_amdgcn_wmma_f32_16x16x32_bf16(false, aq, false, bk0, (short)0, z, false, false);
    v8f s1 = __builtin_amdgcn_wmma_f32_16x16x32_bf16(false, aq, false, bk1, (short)0, z, false, false);

    // streaming softmax per query row (elem r -> q = half*8 + r)
#pragma unroll
    for (int r = 0; r < 8; ++r) {
      float mx = fmaxf(s0[r], s1[r]);
#pragma unroll
      for (int msk = 8; msk >= 1; msk >>= 1) mx = fmaxf(mx, __shfl_xor(mx, msk, 32));
      float nm = fmaxf(mrow[r], mx);
      float corr = __expf(mrow[r] - nm);
      mrow[r] = nm;
      float p0v = __expf(s0[r] - nm);
      float p1v = __expf(s1[r] - nm);
      float rs = p0v + p1v;
#pragma unroll
      for (int msk = 8; msk >= 1; msk >>= 1) rs += __shfl_xor(rs, msk, 32);
      lrow[r] = lrow[r] * corr + rs;
      o0[r] *= corr; o1[r] *= corr;
      s0[r] = p0v; s1[r] = p1v;
    }

    // C-layout P -> LDS [q][key], then read back in A-layout (DS in-order per wave)
#pragma unroll
    for (int r = 0; r < 8; ++r) {
      plds[half * 8 + r][l] = s0[r];
      plds[half * 8 + r][16 + l] = s1[r];
    }
    v16bf ap;
#pragma unroll
    for (int e = 0; e < 8; ++e) ap[e] = tobf(plds[l][half * 8 + e]);
#pragma unroll
    for (int e = 0; e < 8; ++e) ap[8 + e] = tobf(plds[l][16 + half * 8 + e]);

    // V fragments: lane = d (l / 16+l), elems = key in the same order as ap
    const float* vb = qkv + ((long)(b * NN + kt + half * 8) * (3 * EE)) + 2 * EE + h * 32 + l;
    v16bf bv0, bv1;
#pragma unroll
    for (int e = 0; e < 8; ++e) {
      bv0[e] = tobf(vb[(long)e * (3 * EE)]); bv0[8 + e] = tobf(vb[(long)(16 + e) * (3 * EE)]);
      bv1[e] = tobf(vb[(long)e * (3 * EE) + 16]); bv1[8 + e] = tobf(vb[(long)(16 + e) * (3 * EE) + 16]);
    }
    o0 = __builtin_amdgcn_wmma_f32_16x16x32_bf16(false, ap, false, bv0, (short)0, o0, false, false);
    o1 = __builtin_amdgcn_wmma_f32_16x16x32_bf16(false, ap, false, bv1, (short)0, o1, false, false);
  }

#pragma unroll
  for (int r = 0; r < 8; ++r) {
    int row = qt * 16 + half * 8 + r;
    float inv = 1.f / lrow[r];
    float* op = o + ((long)(b * NN + row) * EE) + h * 32;
    op[l] = o0[r] * inv;
    op[16 + l] = o1[r] * inv;
  }
}

// mean over tokens: m[b,e]
__global__ void __launch_bounds__(256) k_mean(const float* __restrict__ xc,
                                              float* __restrict__ m) {
  int t = blockIdx.x * blockDim.x + threadIdx.x;   // B*E
  if (t >= BB * EE) return;
  int b = t >> 8, e = t & (EE - 1);
  const float* p = xc + (long)b * NN * EE + e;
  float s = 0.f;
  for (int n = 0; n < NN; ++n) s += p[(long)n * EE];
  m[t] = s * (1.f / (float)NN);
}

__global__ void __launch_bounds__(256) k_zero(float* p, int n) {
  int t = blockIdx.x * blockDim.x + threadIdx.x;
  if (t < n) p[t] = 0.f;
}

// =====================================================================
// host-side orchestration
// =====================================================================
extern "C" void kernel_launch(void* const* d_in, const int* in_sizes, int n_in,
                              void* d_out, int out_size, void* d_ws, size_t ws_size,
                              hipStream_t stream) {
  const float* x      = (const float*)d_in[0];
  const float* W1     = (const float*)d_in[1];
  const float* b1     = (const float*)d_in[2];
  const float* g1     = (const float*)d_in[3];
  const float* be1    = (const float*)d_in[4];
  const float* W2     = (const float*)d_in[5];
  const float* b2     = (const float*)d_in[6];
  const float* g2     = (const float*)d_in[7];
  const float* be2    = (const float*)d_in[8];
  const float* ln1_g  = (const float*)d_in[9];
  const float* ln1_b  = (const float*)d_in[10];
  const float* attn_w = (const float*)d_in[11];
  const float* attn_b = (const float*)d_in[12];
  const float* out_w  = (const float*)d_in[13];
  const float* out_b  = (const float*)d_in[14];
  const float* ln2_g  = (const float*)d_in[15];
  const float* ln2_b  = (const float*)d_in[16];
  const float* ff1_w  = (const float*)d_in[17];
  const float* ff1_b  = (const float*)d_in[18];
  const float* ff2_w  = (const float*)d_in[19];
  const float* ff2_b  = (const float*)d_in[20];
  const float* dec1_w = (const float*)d_in[21];
  const float* dec1_b = (const float*)d_in[22];
  const float* dec2_w = (const float*)d_in[23];
  const float* dec2_b = (const float*)d_in[24];

  // workspace layout (floats); lifetime-based aliasing, total 32M floats = 128 MiB
  const size_t SZ = (size_t)BN * EE;          // 4,194,304
  float* ws   = (float*)d_ws;
  float* xcur = ws;                            // [BN,E]
  float* hn   = ws + SZ;                       // [BN,E]
  float* obuf = ws + 2 * SZ;                   // [BN,E]; early: idx + BN stats; late: decoder tmp
  float* qkvb = ws + 3 * SZ;                   // [BN,3E]; early: h1 [BN*K,6]
  float* ffb  = ws + 6 * SZ;                   // [BN,2E]; early: h2 [BN,E]

  int*   idxb = (int*)obuf;                    // BN*K ints
  float* stats = obuf + (size_t)BN * KK;       // 544 floats
  float* sum1 = stats, *sq1 = stats + 8;
  float* sum2 = stats + 16, *sq2 = stats + 272;
  float* h1 = qkvb;
  float* h2 = ffb;
  float* mbuf = obuf;                          // [B,E] (after last attention)
  float* dech = obuf + 4096;                   // [B,2E]

  // ---- ARPE ----
  k_zero<<<3, 256, 0, stream>>>(stats, 544);
  k_knn<<<BN / KNN_BLK, KNN_BLK, 0, stream>>>(x, idxb);
  k_arpe1<<<(BN * KK) / 256, 256, 0, stream>>>(x, idxb, W1, b1, h1);
  k_colstats<<<dim3(64, 1), 64, 0, stream>>>(h1, BN * KK, 6, sum1, sq1);
  k_arpe2<<<BN, 256, 0, stream>>>(h1, sum1, sq1, g1, be1, W2, b2, h2);
  k_colstats<<<dim3(64, 4), 64, 0, stream>>>(h2, BN, EE, sum2, sq2);
  k_arpe3<<<(BN * EE) / 256, 256, 0, stream>>>(h2, sum2, sq2, g2, be2, xcur);

  // ---- transformer blocks ----
  for (int i = 0; i < NBLK; ++i) {
    k_ln<<<BN, 256, 0, stream>>>(xcur, ln1_g + i * EE, ln1_b + i * EE, hn);
    { int tiles = (BN / 16) * (3 * EE / 64);
      k_gemm<0><<<(tiles + 7) / 8, 256, 0, stream>>>(hn, attn_w + (size_t)i * EE * 3 * EE,
                                                     attn_b + i * 3 * EE, nullptr, qkvb,
                                                     BN, EE, 3 * EE); }
    k_attn<<<BB * HH * (NN / 16), 32, 0, stream>>>(qkvb, obuf);
    { int tiles = (BN / 16) * (EE / 64);
      k_gemm<0><<<(tiles + 7) / 8, 256, 0, stream>>>(obuf, out_w + (size_t)i * EE * EE,
                                                     out_b + i * EE, hn /*residual=hn*/, xcur,
                                                     BN, EE, EE); }
    k_ln<<<BN, 256, 0, stream>>>(xcur, ln2_g + i * EE, ln2_b + i * EE, hn);
    { int tiles = (BN / 16) * (2 * EE / 64);
      k_gemm<1><<<(tiles + 7) / 8, 256, 0, stream>>>(hn, ff1_w + (size_t)i * EE * 2 * EE,
                                                     ff1_b + i * 2 * EE, nullptr, ffb,
                                                     BN, EE, 2 * EE); }
    { int tiles = (BN / 16) * (EE / 64);
      k_gemm<0><<<(tiles + 7) / 8, 256, 0, stream>>>(ffb, ff2_w + (size_t)i * 2 * EE * EE,
                                                     ff2_b + i * EE, xcur /*residual*/, xcur,
                                                     BN, 2 * EE, EE); }
  }

  // ---- decoder ----
  k_mean<<<(BB * EE) / 256, 256, 0, stream>>>(xcur, mbuf);
  { int tiles = 1 * (2 * EE / 64);   // M=8 -> one (masked) M-tile
    k_gemm<1><<<(tiles + 7) / 8, 256, 0, stream>>>(mbuf, dec1_w, dec1_b, nullptr, dech,
                                                   BB, EE, 2 * EE); }
  { int tiles = 1 * (3072 / 64);
    k_gemm<0><<<(tiles + 7) / 8, 256, 0, stream>>>(dech, dec2_w, dec2_b, nullptr,
                                                   (float*)d_out, BB, 2 * EE, 3072); }
}
